// FeatureDenoiser_83605833384262
// MI455X (gfx1250) — matrix operand order repaired and checked by reading, not compiled, not run on hardware
//
#include <hip/hip_runtime.h>

// ---------------- problem constants ----------------
static constexpr int B_ = 8;
static constexpr int C_ = 512;
static constexpr int Hh = 64;
static constexpr int Ww = 64;
static constexpr int N_ = Hh * Ww;       // 4096
static constexpr int CT = C_ / 16;       // 32 16-tiles along channel dims
static constexpr int NT = N_ / 16;       // 256 16-tiles along spatial dim

// ---------------- WMMA vector types ----------------
typedef __attribute__((ext_vector_type(8)))  __bf16 v8bf;
typedef __attribute__((ext_vector_type(16))) __bf16 v16bf;
typedef __attribute__((ext_vector_type(8)))  float  v8f;

// A fragment: 16x32 bf16, lane = g*16 + m (m = lane%16 is the row).
// element e<8  -> K = k0 + 8*g + e ; e>=8 -> K = k0 + 16 + 8*g + (e-8)
__device__ inline v16bf load_a_frag(const __bf16* __restrict__ base, int lda,
                                    int k0, int m, int g) {
    const v8bf* p0 = (const v8bf*)(base + (size_t)m * lda + k0 + 8 * g);
    const v8bf* p1 = (const v8bf*)(base + (size_t)m * lda + k0 + 16 + 8 * g);
    v8bf lo = *p0;
    v8bf hi = *p1;
    return __builtin_shufflevector(lo, hi, 0, 1, 2, 3, 4, 5, 6, 7,
                                   8, 9, 10, 11, 12, 13, 14, 15);
}

// B fragment: 32x16 bf16, lane = g*16 + n. element e<8 -> K = k0 + 8*g + e ; e>=8 -> K = k0 + 16 + 8*g + (e-8).
// Memory is K-contiguous per column ("row" indexes N over K-major storage).
__device__ inline v16bf load_b_frag(const __bf16* __restrict__ base, int ldb,
                                    int k0, int row, int g) {
    const __bf16* p = base + (size_t)row * ldb + k0 + 8 * g;
    v8bf lo = *(const v8bf*)p;
    v8bf hi = *(const v8bf*)(p + 16);
    return __builtin_shufflevector(lo, hi, 0, 1, 2, 3, 4, 5, 6, 7,
                                   8, 9, 10, 11, 12, 13, 14, 15);
}

// ---------------- kernel 1: transpose+convert x -> xT (b,n,c) bf16 ---------
__global__ void pack_x_kernel(const float* __restrict__ x,
                              __bf16* __restrict__ xT) {
    __shared__ float tile[16][17];
    const int b  = blockIdx.z;
    const int c0 = blockIdx.y * 16;
    const int n0 = blockIdx.x * 16;
    const int tx = threadIdx.x, ty = threadIdx.y;
    tile[ty][tx] = x[((size_t)(b * C_ + c0 + ty)) * N_ + n0 + tx];
    __syncthreads();
    xT[((size_t)(b * N_ + n0 + ty)) * C_ + c0 + tx] = (__bf16)tile[tx][ty];
}

// ---------------- kernel 2: convert the three 1x1 weights to bf16 ----------
__global__ void pack_w_kernel(const float* __restrict__ w1,
                              const float* __restrict__ w2,
                              const float* __restrict__ w3,
                              __bf16* __restrict__ wq) {
    const int i = blockIdx.x * 256 + threadIdx.x;      // 0 .. C*C-1
    wq[i]               = (__bf16)w1[i];
    wq[C_ * C_ + i]     = (__bf16)w2[i];
    wq[2 * C_ * C_ + i] = (__bf16)w3[i];
}

// ======================================================================
// GEMM core: each wave computes a 32(M) x 64(N) macro-tile with 2x4
// register blocking: 2 A fragments + 4 B fragments -> 8 WMMAs per K-step.
// ======================================================================

// ---------------- kernel 3: fused QKV GEMM ---------------------------------
// Y[o,n] = sum_c W[o,c] * x[c,n] + bias[o]
// A = W (o,c) row-major bf16 ; B = xT (n,c) K-contiguous bf16
// sel 0 -> q (b,o,n), sel 1 -> k (b,o,n), sel 2 -> vT (b,n,o)
__global__ void qkv_gemm_kernel(const __bf16* __restrict__ xT,
                                const __bf16* __restrict__ wqkv,
                                const float* __restrict__ b1,
                                const float* __restrict__ b2,
                                const float* __restrict__ b3,
                                __bf16* __restrict__ q,
                                __bf16* __restrict__ k,
                                __bf16* __restrict__ vT) {
    const int wave = threadIdx.x >> 5;
    const int lane = threadIdx.x & 31;
    const int g    = lane >> 4;
    const int nl   = lane & 15;
    const int ng   = blockIdx.x;                  // 64-col group: 0..63
    const int mb   = blockIdx.y * 8 + wave;       // 32-row block: 0..15
    const int bz   = blockIdx.z;
    const int bat  = bz % B_;
    const int sel  = bz / B_;

    const __bf16* A0 = wqkv + (size_t)sel * C_ * C_ + (size_t)(mb * 32) * C_;
    const __bf16* A1 = A0 + (size_t)16 * C_;
    const __bf16* X  = xT + (size_t)bat * N_ * C_ + (size_t)(ng * 64) * C_;

    v8f acc[2][4] = {};
    for (int k0 = 0; k0 < C_; k0 += 32) {
        v16bf a0 = load_a_frag(A0, C_, k0, nl, g);
        v16bf a1 = load_a_frag(A1, C_, k0, nl, g);
        v16bf bf[4];
#pragma unroll
        for (int j = 0; j < 4; ++j)
            bf[j] = load_b_frag(X, C_, k0, j * 16 + nl, g);
#pragma unroll
        for (int j = 0; j < 4; ++j) {
            acc[0][j] = __builtin_amdgcn_wmma_f32_16x16x32_bf16(
                false, a0, false, bf[j], (short)0, acc[0][j], false, false);
            acc[1][j] = __builtin_amdgcn_wmma_f32_16x16x32_bf16(
                false, a1, false, bf[j], (short)0, acc[1][j], false, false);
        }
    }

    const float* bias = (sel == 0) ? b1 : (sel == 1) ? b2 : b3;
#pragma unroll
    for (int mt = 0; mt < 2; ++mt) {
#pragma unroll
        for (int r = 0; r < 8; ++r) {
            const int o = mb * 32 + mt * 16 + 8 * g + r;
            const float bo = bias[o];
#pragma unroll
            for (int j = 0; j < 4; ++j) {
                const int n = ng * 64 + j * 16 + nl;
                const float y = acc[mt][j][r] + bo;
                if (sel == 0)
                    q[((size_t)(bat * C_ + o)) * N_ + n] = (__bf16)y;
                else if (sel == 1)
                    k[((size_t)(bat * C_ + o)) * N_ + n] = (__bf16)y;
                else
                    vT[((size_t)(bat * N_ + n)) * C_ + o] = (__bf16)y;
            }
        }
    }
}

// ---------------- kernel 4: scores = q . k^T (K = N = 4096) ----------------
__global__ void scores_gemm_kernel(const __bf16* __restrict__ q,
                                   const __bf16* __restrict__ k,
                                   float* __restrict__ scores) {
    const int wave = threadIdx.x >> 5;
    const int lane = threadIdx.x & 31;
    const int g    = lane >> 4;
    const int nl   = lane & 15;
    const int dg   = blockIdx.x;                  // 64-col group of d: 0..7
    const int mb   = blockIdx.y * 8 + wave;       // 32-row block of c: 0..15
    const int bat  = blockIdx.z;

    const __bf16* A0 = q + (size_t)bat * C_ * N_ + (size_t)(mb * 32) * N_;
    const __bf16* A1 = A0 + (size_t)16 * N_;
    const __bf16* Kp = k + (size_t)bat * C_ * N_ + (size_t)(dg * 64) * N_;

    v8f acc[2][4] = {};
    for (int k0 = 0; k0 < N_; k0 += 32) {
        if (k0 + 512 < N_) {  // uniform: EXEC stays all-ones
            __builtin_prefetch(A0 + (size_t)nl * N_ + k0 + 512, 0, 1);
            __builtin_prefetch(Kp + (size_t)nl * N_ + k0 + 512, 0, 1);
        }
        v16bf a0 = load_a_frag(A0, N_, k0, nl, g);
        v16bf a1 = load_a_frag(A1, N_, k0, nl, g);
        v16bf bf[4];
#pragma unroll
        for (int j = 0; j < 4; ++j)
            bf[j] = load_b_frag(Kp, N_, k0, j * 16 + nl, g);
#pragma unroll
        for (int j = 0; j < 4; ++j) {
            acc[0][j] = __builtin_amdgcn_wmma_f32_16x16x32_bf16(
                false, a0, false, bf[j], (short)0, acc[0][j], false, false);
            acc[1][j] = __builtin_amdgcn_wmma_f32_16x16x32_bf16(
                false, a1, false, bf[j], (short)0, acc[1][j], false, false);
        }
    }

#pragma unroll
    for (int mt = 0; mt < 2; ++mt) {
#pragma unroll
        for (int r = 0; r < 8; ++r) {
            const int c = mb * 32 + mt * 16 + 8 * g + r;
#pragma unroll
            for (int j = 0; j < 4; ++j) {
                const int d = dg * 64 + j * 16 + nl;
                scores[((size_t)(bat * C_ + c)) * C_ + d] = acc[mt][j][r];
            }
        }
    }
}

// ---------------- kernel 5: row softmax -> bf16 attn -----------------------
__global__ void softmax_kernel(const float* __restrict__ scores,
                               __bf16* __restrict__ attn) {
    const int row = blockIdx.x;          // 0 .. B*C-1
    const int t   = threadIdx.x;         // 256 threads, 2 elements each
    const float* s = scores + (size_t)row * C_;
    __shared__ float red[256];

    const float v0 = s[t];
    const float v1 = s[t + 256];

    red[t] = fmaxf(v0, v1);
    __syncthreads();
    for (int off = 128; off > 0; off >>= 1) {
        if (t < off) red[t] = fmaxf(red[t], red[t + off]);
        __syncthreads();
    }
    const float m = red[0];
    __syncthreads();

    const float e0 = __expf(v0 - m);
    const float e1 = __expf(v1 - m);
    red[t] = e0 + e1;
    __syncthreads();
    for (int off = 128; off > 0; off >>= 1) {
        if (t < off) red[t] += red[t + off];
        __syncthreads();
    }
    const float inv = 1.0f / red[0];

    attn[(size_t)row * C_ + t]       = (__bf16)(e0 * inv);
    attn[(size_t)row * C_ + t + 256] = (__bf16)(e1 * inv);
}

// ---------------- kernel 6: denoised = attn . v, fused epilogue ------------
// out = x + 0.1*denoised + 0.1*(depthwise3x3(x) + bg)
__global__ void attnv_fused_kernel(const __bf16* __restrict__ attn,
                                   const __bf16* __restrict__ vT,
                                   const float* __restrict__ x,
                                   const float* __restrict__ wg,
                                   const float* __restrict__ bg,
                                   float* __restrict__ out) {
    const int wave = threadIdx.x >> 5;
    const int lane = threadIdx.x & 31;
    const int g    = lane >> 4;
    const int nl   = lane & 15;
    const int ng   = blockIdx.x;                  // 64-col group of n: 0..63
    const int mb   = blockIdx.y * 8 + wave;       // 32-row block of c: 0..15
    const int bat  = blockIdx.z;

    const __bf16* A0 = attn + (size_t)bat * C_ * C_ + (size_t)(mb * 32) * C_;
    const __bf16* A1 = A0 + (size_t)16 * C_;
    const __bf16* Vp = vT + (size_t)bat * N_ * C_ + (size_t)(ng * 64) * C_;

    v8f acc[2][4] = {};
    for (int k0 = 0; k0 < C_; k0 += 32) {
        v16bf a0 = load_a_frag(A0, C_, k0, nl, g);
        v16bf a1 = load_a_frag(A1, C_, k0, nl, g);
        v16bf bf[4];
#pragma unroll
        for (int j = 0; j < 4; ++j)
            bf[j] = load_b_frag(Vp, C_, k0, j * 16 + nl, g);
#pragma unroll
        for (int j = 0; j < 4; ++j) {
            acc[0][j] = __builtin_amdgcn_wmma_f32_16x16x32_bf16(
                false, a0, false, bf[j], (short)0, acc[0][j], false, false);
            acc[1][j] = __builtin_amdgcn_wmma_f32_16x16x32_bf16(
                false, a1, false, bf[j], (short)0, acc[1][j], false, false);
        }
    }

#pragma unroll
    for (int mt = 0; mt < 2; ++mt) {
#pragma unroll
        for (int r = 0; r < 8; ++r) {
            const int c = mb * 32 + mt * 16 + 8 * g + r;
            const float* xc = x + ((size_t)(bat * C_ + c)) * N_;
            float wloc[9];
#pragma unroll
            for (int t9 = 0; t9 < 9; ++t9) wloc[t9] = wg[c * 9 + t9];
            const float bgc = bg[c];
#pragma unroll
            for (int j = 0; j < 4; ++j) {
                const int n = ng * 64 + j * 16 + nl;
                const int h = n >> 6;
                const int w = n & 63;
                float local = bgc;
#pragma unroll
                for (int dh = -1; dh <= 1; ++dh) {
#pragma unroll
                    for (int dw = -1; dw <= 1; ++dw) {
                        const int hh = h + dh;
                        const int ww = w + dw;
                        float xv = 0.0f;
                        if (hh >= 0 && hh < Hh && ww >= 0 && ww < Ww)
                            xv = xc[hh * Ww + ww];
                        local += wloc[(dh + 1) * 3 + (dw + 1)] * xv;
                    }
                }
                out[((size_t)(bat * C_ + c)) * N_ + n] =
                    xc[n] + 0.1f * acc[mt][j][r] + 0.1f * local;
            }
        }
    }
}

// ---------------- host launcher --------------------------------------------
extern "C" void kernel_launch(void* const* d_in, const int* in_sizes, int n_in,
                              void* d_out, int out_size, void* d_ws, size_t ws_size,
                              hipStream_t stream) {
    const float* x  = (const float*)d_in[0];
    const float* w1 = (const float*)d_in[1];
    const float* b1 = (const float*)d_in[2];
    const float* w2 = (const float*)d_in[3];
    const float* b2 = (const float*)d_in[4];
    const float* w3 = (const float*)d_in[5];
    const float* b3 = (const float*)d_in[6];
    const float* wg = (const float*)d_in[7];
    const float* bg = (const float*)d_in[8];
    float* out = (float*)d_out;

    // carve workspace (all blocks 16B-aligned by construction)
    const size_t bcn = (size_t)B_ * C_ * N_;   // 16.7M elements
    __bf16* xT    = (__bf16*)d_ws;             // (b,n,c) bf16
    __bf16* wqkv  = xT + bcn;                  // 3*C*C bf16
    __bf16* qb    = wqkv + (size_t)3 * C_ * C_;// (b,c,n) bf16
    __bf16* kb    = qb + bcn;                  // (b,c,n) bf16
    __bf16* vTb   = kb + bcn;                  // (b,n,c) bf16
    float*  sc    = (float*)(vTb + bcn);       // (b,c,c) f32
    __bf16* attnb = (__bf16*)(sc + (size_t)B_ * C_ * C_); // (b,c,c) bf16

    pack_x_kernel<<<dim3(N_ / 16, C_ / 16, B_), dim3(16, 16), 0, stream>>>(x, xT);
    pack_w_kernel<<<dim3((C_ * C_) / 256), dim3(256), 0, stream>>>(w1, w2, w3, wqkv);
    // waves: 32x64 macro-tiles; block = 8 waves sharing the B-tile stream
    qkv_gemm_kernel<<<dim3(NT / 4, 2, B_ * 3), dim3(256), 0, stream>>>(
        xT, wqkv, b1, b2, b3, qb, kb, vTb);
    scores_gemm_kernel<<<dim3(C_ / 64, 2, B_), dim3(256), 0, stream>>>(qb, kb, sc);
    softmax_kernel<<<dim3(B_ * C_), dim3(256), 0, stream>>>(sc, attnb);
    attnv_fused_kernel<<<dim3(NT / 4, 2, B_), dim3(256), 0, stream>>>(
        attnb, vTb, x, wg, bg, out);
}
